// RGCNLayer_10342281248977
// MI455X (gfx1250) — matrix operand order repaired and checked by reading, not compiled, not run on hardware
//
#include <hip/hip_runtime.h>
#include <hip/hip_bf16.h>

typedef __attribute__((ext_vector_type(16))) __bf16 v16bf;
typedef __attribute__((ext_vector_type(8)))  float  v8f;

#define IN_DIM   128
#define OUT_DIM  128
#define NREL     8
#define NBASES   4
#define KTOT     (IN_DIM + NREL * IN_DIM)     // 1152
#define KBLOCKS  (KTOT / 32)                  // 36
#define NSTAGES  (KTOT / 128)                 // 9 stages of 128 K-columns
#define NTILES   (OUT_DIM / 16)               // 8
#define WP_DWORDS (KBLOCKS * NTILES * 32 * 8) // 73728 packed bf16x2

static __device__ __forceinline__ unsigned short f32_to_bf16_bits(float f) {
    __bf16 h = (__bf16)f;
    return __builtin_bit_cast(unsigned short, h);
}

// CDNA5 async global->LDS copy (ASYNCcnt-tracked, bypasses VGPRs).
// GVS mode: mem_addr = SGPR base + 32-bit VGPR byte offset.
static __device__ __forceinline__ void async_copy_b128(unsigned lds_addr,
                                                       const void* gbase,
                                                       unsigned goff) {
    asm volatile("global_load_async_to_lds_b128 %0, %1, %2"
                 :
                 : "v"(lds_addr), "v"(goff), "s"(gbase)
                 : "memory");
}

static __device__ __forceinline__ void wait_async_zero() {
    asm volatile("s_wait_asynccnt 0x0" ::: "memory");
}

// ---------------------------------------------------------------------------
// Build concatenated weight [W_self ; W_rel(r=0..7)] (KTOT x 128), where
// W_rel[r] = sum_b coeff[r,b] * bases[b], stored pre-swizzled in the
// B order this code uses for v_wmma_f32_16x16x32_bf16 (contiguous k, as A) so GEMM loads are coalesced.
// dword index = ((kb*8 + ntile)*32 + lane)*8 + v
//   lane -> column N = ntile*16 + (lane&15); Kbase = kb*32 + (lane<16 ? 0:16)
//   dword v holds bf16 pair (K = Kbase+2v, Kbase+2v+1)
// ---------------------------------------------------------------------------
__global__ void rgcn_pack_weights(const float* __restrict__ Wself,
                                  const float* __restrict__ bases,
                                  const float* __restrict__ coeff,
                                  unsigned int* __restrict__ Wp) {
    int t = blockIdx.x * blockDim.x + threadIdx.x;
    if (t >= WP_DWORDS) return;
    int v    = t & 7;
    int lane = (t >> 3) & 31;
    int nt   = (t >> 8) & 7;
    int kb   = t >> 11;

    int n  = nt * 16 + (lane & 15);
    int Kb = kb * 32 + ((lane & 16) ? 16 : 0);

    float w[2];
#pragma unroll
    for (int j = 0; j < 2; ++j) {
        int k = Kb + 2 * v + j;
        float val;
        if (k < IN_DIM) {
            val = Wself[k * OUT_DIM + n];
        } else {
            int r = (k - IN_DIM) >> 7;
            int i = (k - IN_DIM) & 127;
            val = 0.f;
#pragma unroll
            for (int b = 0; b < NBASES; ++b)
                val += coeff[r * NBASES + b] *
                       bases[((size_t)b * IN_DIM + i) * OUT_DIM + n];
        }
        w[j] = val;
    }
    unsigned int lo = f32_to_bf16_bits(w[0]);
    unsigned int hi = f32_to_bf16_bits(w[1]);
    Wp[t] = lo | (hi << 16);
}

// ---------------------------------------------------------------------------
// Edge scatter: agg[(tgt*8 + rel)*128 + :] += X[src, :]
// One wave32 per edge, float4 per lane, hardware fp32 atomics.
// ---------------------------------------------------------------------------
__global__ void rgcn_scatter(const float* __restrict__ X,
                             const int* __restrict__ eidx,
                             const int* __restrict__ etype,
                             float* __restrict__ Agg,
                             int nEdges) {
    int edge = blockIdx.x * (blockDim.x >> 5) + (threadIdx.x >> 5);
    if (edge >= nEdges) return;
    int lane = threadIdx.x & 31;

    int src = eidx[edge];
    int tgt = eidx[nEdges + edge];
    int rel = etype[edge];

    const float4* xs = (const float4*)(X + (size_t)src * IN_DIM);
    float* dst = Agg + ((size_t)tgt * NREL + rel) * IN_DIM + lane * 4;
    float4 val = xs[lane];
    unsafeAtomicAdd(dst + 0, val.x);
    unsafeAtomicAdd(dst + 1, val.y);
    unsafeAtomicAdd(dst + 2, val.z);
    unsafeAtomicAdd(dst + 3, val.w);
}

// ---------------------------------------------------------------------------
// Fused GEMM: out[n,:] = relu([X | agg][n,:] @ Wcat + b)
// Block = 16 nodes, 8 waves = 8 x 16-column output tiles.
// A staged in 128-K-column panels via async global->LDS, double buffered.
// ---------------------------------------------------------------------------
__global__ void __launch_bounds__(256)
rgcn_gemm(const float* __restrict__ X,
          const float* __restrict__ Agg,
          const uint4* __restrict__ Wp,
          const float* __restrict__ bias,
          float* __restrict__ out,
          int nNodes) {
    __shared__ float As[2][16 * 128];   // 2 x 8KB double-buffered A panel

    int tid  = threadIdx.x;
    int lane = tid & 31;
    int wave = tid >> 5;            // output column tile 0..7
    int nodeBase = blockIdx.x * 16;
    int laneRow  = lane & 15;

    v8f acc = {0.f, 0.f, 0.f, 0.f, 0.f, 0.f, 0.f, 0.f};

    // Per-thread async-copy work items: 512 x b128 per stage, 2 per thread.
    // e = tid + 256*j ; row = e>>4 (16 x 16B per 512B row), c16 = e&15.
    int r0 = tid >> 4;              // j=0 row  (0..15)
    int c0 = (tid & 15) * 16;       // j=0 byte column in row
    int r1 = (tid + 256) >> 4;      // j=1 row
    int c1 = c0;                    // same column pattern
    int grow0 = nodeBase + r0; if (grow0 >= nNodes) grow0 = nNodes - 1;
    int grow1 = nodeBase + r1; if (grow1 >= nNodes) grow1 = nNodes - 1;

    unsigned ldsBase[2];
    ldsBase[0] = (unsigned)(uintptr_t)&As[0][0];
    ldsBase[1] = (unsigned)(uintptr_t)&As[1][0];

    // Issue stage s (s=0 -> X, s>=1 -> Agg relation s-1) into buffer buf.
    auto issue_stage = [&](int s, int buf) {
        unsigned l0 = ldsBase[buf] + (unsigned)(r0 * 512 + c0);
        unsigned l1 = ldsBase[buf] + (unsigned)(r1 * 512 + c1);
        if (s == 0) {
            async_copy_b128(l0, X, (unsigned)(grow0 * 512 + c0));
            async_copy_b128(l1, X, (unsigned)(grow1 * 512 + c1));
        } else {
            unsigned rel = (unsigned)(s - 1) * 512u;
            async_copy_b128(l0, Agg, (unsigned)grow0 * 4096u + rel + (unsigned)c0);
            async_copy_b128(l1, Agg, (unsigned)grow1 * 4096u + rel + (unsigned)c1);
        }
    };

    issue_stage(0, 0);

    for (int s = 0; s < NSTAGES; ++s) {
        int buf = s & 1;
        wait_async_zero();          // my stage-s copies have landed in LDS
        __syncthreads();            // everyone's stage-s copies visible;
                                    // also: all waves done reading buf^1
        if (s + 1 < NSTAGES) issue_stage(s + 1, buf ^ 1);

        const float* Ab = &As[buf][0];
        int b0 = (lane & 16) ? 16 : 0;
#pragma unroll
        for (int kk = 0; kk < 4; ++kk) {
            const float* ls = Ab + laneRow * 128 + kk * 32 + b0;
            v16bf a;
#pragma unroll
            for (int j = 0; j < 8; ++j) a[j]     = (__bf16)ls[j];
#pragma unroll
            for (int j = 0; j < 8; ++j) a[8 + j] = (__bf16)ls[8 + j];

            int kb = s * 4 + kk;
            const uint4* wp = Wp + (((size_t)kb * NTILES + wave) * 32 + lane) * 2;
            union { uint4 q[2]; v16bf v; } ub;
            ub.q[0] = wp[0];
            ub.q[1] = wp[1];

            acc = __builtin_amdgcn_wmma_f32_16x16x32_bf16(
                false, a, false, ub.v, (short)0, acc, false, false);
        }
    }

    // Epilogue: C/D layout -> lane gives N, VGPR v gives M (lanes>=16: M+8)
    int col  = wave * 16 + laneRow;
    float bv = bias[col];
    int mBase = (lane & 16) ? 8 : 0;
#pragma unroll
    for (int v = 0; v < 8; ++v) {
        int row = nodeBase + mBase + v;
        if (row < nNodes) {
            float x = acc[v] + bv;
            out[(size_t)row * OUT_DIM + col] = x > 0.f ? x : 0.f;
        }
    }
}

// ---------------------------------------------------------------------------
extern "C" void kernel_launch(void* const* d_in, const int* in_sizes, int n_in,
                              void* d_out, int out_size, void* d_ws, size_t ws_size,
                              hipStream_t stream) {
    const float* X      = (const float*)d_in[0];
    const int*   eidx   = (const int*)d_in[1];
    const int*   etype  = (const int*)d_in[2];
    const float* Wself  = (const float*)d_in[3];
    const float* bself  = (const float*)d_in[4];
    const float* bases  = (const float*)d_in[5];
    const float* coeff  = (const float*)d_in[6];
    float*       out    = (float*)d_out;

    int nNodes = in_sizes[0] / IN_DIM;
    int nEdges = in_sizes[2];

    size_t aggBytes = (size_t)nNodes * NREL * IN_DIM * sizeof(float);
    float*        Agg = (float*)d_ws;
    unsigned int* Wp  = (unsigned int*)((char*)d_ws + aggBytes);

    hipMemsetAsync(Agg, 0, aggBytes, stream);

    rgcn_pack_weights<<<(WP_DWORDS + 255) / 256, 256, 0, stream>>>(
        Wself, bases, coeff, Wp);

    rgcn_scatter<<<(nEdges + 7) / 8, 256, 0, stream>>>(
        X, eidx, etype, Agg, nEdges);

    rgcn_gemm<<<(nNodes + 15) / 16, 256, 0, stream>>>(
        X, Agg, (const uint4*)Wp, bself, out, nNodes);
}
